// TemporalTokenizer_1159641170098
// MI455X (gfx1250) — matrix operand order repaired and checked by reading, not compiled, not run on hardware
//
#include <hip/hip_runtime.h>
#include <math.h>

// Problem constants (match reference)
#define BATCH   32
#define SLEN    4096
#define DIM     512
#define TOKEN   16
#define WSTRIDE 8
#define NWIN    511                  // (4096-16)/8+1
#define MTOT    (BATCH * NWIN)       // 16352
#define KTOT    (TOKEN * DIM)        // 8192

// GEMM tiling
#define MT 128
#define NT 128
#define KT 64                        // K per LDS stage: 16 WMMAs between barriers
#define NKSTEP (KTOT / KT)           // 128
#define LDT 88                       // KT + 24 pad (176B rows: 16B-aligned, bank period 16)

typedef _Float16 half_t;
typedef __attribute__((ext_vector_type(16))) _Float16 v16h;
typedef __attribute__((ext_vector_type(8)))  _Float16 v8h;
typedef __attribute__((ext_vector_type(4)))  _Float16 v4h;
typedef __attribute__((ext_vector_type(8)))  float    v8f;
typedef __attribute__((ext_vector_type(4)))  float    v4f;

static __device__ __forceinline__ v16h mk16(v8h lo, v8h hi) {
    union { v16h v; v8h h[2]; } u;
    u.h[0] = lo; u.h[1] = hi;
    return u.v;
}

// ---------------------------------------------------------------------------
// Kernel 1: windowed GEMM (M=16352, K=8192, N=512) + bias + exact GELU.
// A row m = b*NWIN + n is the contiguous slice x[(b*SLEN + n*WSTRIDE)*DIM + k]
// (since (n*8 + t)*512 + c == n*4096 + k for k = t*512+c), so the sliding
// window never needs materializing. f32 -> f16 in registers, f32 accumulation
// via V_WMMA_F32_16X16X32_F16. h is staged into d_out for kernel 2.
//
// Out-of-range M rows (last block only) load from a clamped in-bounds address
// and compute garbage that the epilogue's row guard discards — no per-element
// zeroing needed in the hot loop.
// ---------------------------------------------------------------------------
__global__ __launch_bounds__(256) void tt_gemm_gelu(
    const float* __restrict__ x, const float* __restrict__ W,
    const float* __restrict__ bvec, float* __restrict__ out)
{
    __shared__ __align__(16) half_t Ash[2][MT][LDT];   // A tile, row-major
    __shared__ __align__(16) half_t Bsh[2][NT][LDT];   // B tile, K-transposed

    const int tid  = threadIdx.x;
    const int lane = tid & 31;
    const int wid  = tid >> 5;            // 8 waves
    const int m0   = blockIdx.y * MT;
    const int n0   = blockIdx.x * NT;

    // ---- A global-load mapping: 8 threads/row (2 col-halves of 32), 32 rows/pass, 4 passes
    const int arow = tid >> 3;            // 0..31
    const int acol = (tid & 7) * 4;       // 0..28
    const float* aptr[4];
#pragma unroll
    for (int p = 0; p < 4; ++p) {
        int m  = m0 + p * 32 + arow;
        int mc = (m < MTOT) ? m : 0;      // clamp: always in-bounds; garbage
        int b  = mc / NWIN;               // rows are discarded by the epilogue
        int n  = mc - b * NWIN;
        aptr[p] = x + (size_t)(b * SLEN + n * WSTRIDE) * DIM + acol;
    }

    // ---- B global-load mapping: column bnc, 32 consecutive K rows per thread
    const int bnc = tid & 127;            // 0..127  (column inside tile)
    const int bkr = (tid >> 7) * 32;      // 0 or 32 (K base inside tile)
    const float* wptr = W + (size_t)bkr * DIM + (n0 + bnc);

    v4f   areg[4][2];                     // 4 passes x 2 col-halves
    float breg[32];

    auto loadTiles = [&](int KK) {
#pragma unroll
        for (int p = 0; p < 4; ++p) {
            areg[p][0] = *(const v4f*)(aptr[p] + KK);
            areg[p][1] = *(const v4f*)(aptr[p] + KK + 32);
        }
        const float* wp = wptr + (size_t)KK * DIM;
#pragma unroll
        for (int j = 0; j < 32; ++j)
            breg[j] = wp[(size_t)j * DIM];
    };
    auto storeTiles = [&](int buf) {
#pragma unroll
        for (int p = 0; p < 4; ++p) {
#pragma unroll
            for (int h = 0; h < 2; ++h) {
                v4h hv;
#pragma unroll
                for (int e = 0; e < 4; ++e) hv[e] = (half_t)areg[p][h][e];
                *(v4h*)&Ash[buf][p * 32 + arow][acol + h * 32] = hv;
            }
        }
#pragma unroll
        for (int q = 0; q < 4; ++q) {
            v8h hv;
#pragma unroll
            for (int e = 0; e < 8; ++e) hv[e] = (half_t)breg[q * 8 + e];
            *(v8h*)&Bsh[buf][bnc][bkr + q * 8] = hv;
        }
    };

    // ---- wave -> output sub-tile mapping: 4 M-waves x 2 N-waves, 32x64 each
    const int waveM = wid & 3;
    const int waveN = wid >> 2;
    const int fragRow = (lane & 15);
    const int k0a = (lane < 16) ? 0 : 8;   // ISA 16-bit A layout (16x32)
    const int k0b = (lane < 16) ? 0 : 8;   // B: same K layout as A (16x32 table order)

    v8f c[2][4];
#pragma unroll
    for (int i = 0; i < 2; ++i)
#pragma unroll
        for (int j = 0; j < 4; ++j) {
            v8f z = {0.f, 0.f, 0.f, 0.f, 0.f, 0.f, 0.f, 0.f};
            c[i][j] = z;
        }

    loadTiles(0);
    storeTiles(0);
    __syncthreads();

    int buf = 0;
    for (int ks = 0; ks < NKSTEP; ++ks) {
        if (ks + 1 < NKSTEP) loadTiles((ks + 1) * KT);   // prefetch to regs

#pragma unroll
        for (int s = 0; s < 2; ++s) {                    // two K=32 sub-steps
            v16h aF[2], bF[4];
#pragma unroll
            for (int i = 0; i < 2; ++i) {
                const half_t* ap =
                    &Ash[buf][waveM * 32 + i * 16 + fragRow][s * 32 + k0a];
                aF[i] = mk16(*(const v8h*)ap, *(const v8h*)(ap + 16));
            }
#pragma unroll
            for (int j = 0; j < 4; ++j) {
                const half_t* bp =
                    &Bsh[buf][waveN * 64 + j * 16 + fragRow][s * 32 + k0b];
                bF[j] = mk16(*(const v8h*)bp, *(const v8h*)(bp + 16));
            }
#pragma unroll
            for (int i = 0; i < 2; ++i)
#pragma unroll
                for (int j = 0; j < 4; ++j)
                    c[i][j] = __builtin_amdgcn_wmma_f32_16x16x32_f16(
                        false, aF[i], false, bF[j], (short)0, c[i][j],
                        false, false);
        }

        __syncthreads();
        if (ks + 1 < NKSTEP) {
            storeTiles(buf ^ 1);
            buf ^= 1;
            __syncthreads();
        }
    }

    // ---- epilogue: bias + exact GELU, stage h into out
    const int rbase = (lane < 16) ? 0 : 8;   // C layout: lanes 16-31 -> M+8
#pragma unroll
    for (int i = 0; i < 2; ++i)
#pragma unroll
        for (int j = 0; j < 4; ++j) {
            int col = n0 + waveN * 64 + j * 16 + fragRow;
            float bv = bvec[col];
#pragma unroll
            for (int r = 0; r < 8; ++r) {
                int row = m0 + waveM * 32 + i * 16 + rbase + r;
                if (row < MTOT) {
                    float v = c[i][j][r] + bv;
                    float g = 0.5f * v * (1.0f + erff(v * 0.70710678118654752f));
                    out[(size_t)row * DIM + col] = g;
                }
            }
        }
}

// ---------------------------------------------------------------------------
// Kernel 2: in-place LayerNorm + affine. One wave32 per row of 512 floats.
// ---------------------------------------------------------------------------
__global__ __launch_bounds__(256) void tt_layernorm(
    float* __restrict__ h, const float* __restrict__ gamma,
    const float* __restrict__ beta)
{
    const int lane = threadIdx.x & 31;
    const int wid  = threadIdx.x >> 5;
    const int row  = blockIdx.x * 8 + wid;
    if (row >= MTOT) return;

    float* p = h + (size_t)row * DIM;
    v4f vals[4];
    float s = 0.f, s2 = 0.f;
#pragma unroll
    for (int k = 0; k < 4; ++k) {
        vals[k] = *(const v4f*)(p + (size_t)(lane + 32 * k) * 4);
#pragma unroll
        for (int e = 0; e < 4; ++e) {
            float v = vals[k][e];
            s += v;
            s2 = fmaf(v, v, s2);
        }
    }
#pragma unroll
    for (int off = 16; off > 0; off >>= 1) {
        s  += __shfl_xor(s,  off, 32);
        s2 += __shfl_xor(s2, off, 32);
    }
    const float mu  = s * (1.0f / DIM);
    const float var = s2 * (1.0f / DIM) - mu * mu;
    const float inv = rsqrtf(var + 1e-5f);

#pragma unroll
    for (int k = 0; k < 4; ++k) {
        size_t cb = (size_t)(lane + 32 * k) * 4;
        v4f g  = *(const v4f*)(gamma + cb);
        v4f bt = *(const v4f*)(beta + cb);
        v4f o;
#pragma unroll
        for (int e = 0; e < 4; ++e)
            o[e] = (vals[k][e] - mu) * inv * g[e] + bt[e];
        *(v4f*)(p + cb) = o;
    }
}

extern "C" void kernel_launch(void* const* d_in, const int* in_sizes, int n_in,
                              void* d_out, int out_size, void* d_ws, size_t ws_size,
                              hipStream_t stream)
{
    const float* x     = (const float*)d_in[0];
    const float* W     = (const float*)d_in[1];
    const float* b     = (const float*)d_in[2];
    const float* gamma = (const float*)d_in[3];
    const float* beta  = (const float*)d_in[4];
    float* out = (float*)d_out;

    dim3 grid(DIM / NT, (MTOT + MT - 1) / MT);   // (4, 128) = 512 workgroups
    tt_gemm_gelu<<<grid, 256, 0, stream>>>(x, W, b, out);

    tt_layernorm<<<(MTOT + 7) / 8, 256, 0, stream>>>(out, gamma, beta);
}
